// NGF_40699110097036
// MI455X (gfx1250) — compile-verified, matrix operand order checked by reading, not run on hardware
//
#include <hip/hip_runtime.h>

// ---------------------------------------------------------------------------
// GCN autoencoder for MI455X (gfx1250, wave32, WMMA bf16 16x16x32).
//   z   = adj @ relu(adj @ (x@W1)) @ W2          (dominant: 2 passes over 1GiB adj)
//   out = sigmoid( relu(zi)@weff[:128] + relu(zj)@weff[128:] + (zi*zj)@Wthree[128:] )
// weff = Wtwo @ Wthree[:128]  (exact algebraic fold of the decoder GEMM).
// adj streamed fp32 -> LDS (global_load_async_to_lds_b128, double-buffered),
// converted with v_cvt_pk_bf16_f32; 64-row M-tiles give 4x B-operand register
// reuse so L2 B-traffic (2 GiB) no longer dominates HBM A-traffic (1 GiB/pass).
// ---------------------------------------------------------------------------

typedef __bf16 bf16_t;
typedef __attribute__((ext_vector_type(16))) __bf16 v16bf;
typedef __attribute__((ext_vector_type(8)))  __bf16 v8bf;
typedef __attribute__((ext_vector_type(2)))  __bf16 v2bf;
typedef __attribute__((ext_vector_type(8)))  float  v8f;
typedef __attribute__((ext_vector_type(4)))  float  v4f;

#define WMMA_BF16(a, b, c) \
  __builtin_amdgcn_wmma_f32_16x16x32_bf16(false, (a), false, (b), (short)0, (c), false, false)

#define NNODES 16384
#define INDIM  512
#define H1DIM  256
#define H2DIM  128
#define KCH    64   // adj K-chunk staged in LDS per step
#define MROWS  64   // M rows per block in the adj GEMMs

// -------- async global->LDS staging (CDNA5 path, guarded) -------------------
#if defined(__HIP_DEVICE_COMPILE__) && \
    __has_builtin(__builtin_amdgcn_global_load_async_to_lds_b128)
#define USE_ASYNC_LDS 1
#else
#define USE_ASYNC_LDS 0
#endif

#if USE_ASYNC_LDS
// builtin parameter types (from compiler diagnostic): pointer to
// __attribute__((vector_size(16))) int, global (AS1) src / LDS (AS3) dst.
typedef int gv4i __attribute__((vector_size(16)));
typedef __attribute__((address_space(1))) gv4i as1_v4i;
typedef __attribute__((address_space(3))) gv4i as3_v4i;
#if __has_builtin(__builtin_amdgcn_s_wait_asynccnt)
#define WAIT_ASYNC(n) __builtin_amdgcn_s_wait_asynccnt(n)
#else
#define WAIT_ASYNC(n) asm volatile("s_wait_asynccnt " #n ::: "memory")
#endif
#else
#define WAIT_ASYNC(n)
#endif

// -------- f32 -> bf16 conversion (native pk cvt when available) -------------
__device__ __forceinline__ bf16_t bf(float x) { return (bf16_t)x; }

__device__ __forceinline__ v16bf cvt16(v4f a0, v4f a1, v4f a2, v4f a3) {
  v16bf f;
#if __has_builtin(__builtin_amdgcn_cvt_pk_bf16_f32)
  v2bf p;
  p = __builtin_amdgcn_cvt_pk_bf16_f32(a0[0], a0[1]); f[0] = p[0];  f[1] = p[1];
  p = __builtin_amdgcn_cvt_pk_bf16_f32(a0[2], a0[3]); f[2] = p[0];  f[3] = p[1];
  p = __builtin_amdgcn_cvt_pk_bf16_f32(a1[0], a1[1]); f[4] = p[0];  f[5] = p[1];
  p = __builtin_amdgcn_cvt_pk_bf16_f32(a1[2], a1[3]); f[6] = p[0];  f[7] = p[1];
  p = __builtin_amdgcn_cvt_pk_bf16_f32(a2[0], a2[1]); f[8] = p[0];  f[9] = p[1];
  p = __builtin_amdgcn_cvt_pk_bf16_f32(a2[2], a2[3]); f[10] = p[0]; f[11] = p[1];
  p = __builtin_amdgcn_cvt_pk_bf16_f32(a3[0], a3[1]); f[12] = p[0]; f[13] = p[1];
  p = __builtin_amdgcn_cvt_pk_bf16_f32(a3[2], a3[3]); f[14] = p[0]; f[15] = p[1];
#else
#pragma unroll
  for (int i = 0; i < 4; ++i) {
    f[i] = bf(a0[i]); f[4 + i] = bf(a1[i]); f[8 + i] = bf(a2[i]); f[12 + i] = bf(a3[i]);
  }
#endif
  return f;
}

// -------- fragment loaders (ISA 7.12.2 16-bit A layout) ---------------------
// lanes 0-15 -> rows M=lane, K chunks [0..7],[16..23];
// lanes 16-31 -> rows M=lane-16, K chunks [8..15],[24..31].
__device__ __forceinline__ v16bf load_afrag_f32(const float* __restrict__ base,
                                                long ld, int row0, int k0) {
  int lane = threadIdx.x & 31;
  int r = lane & 15, sel = lane >> 4;
  const float* p = base + (long)(row0 + r) * ld + k0 + sel * 8;
  return cvt16(*(const v4f*)(p + 0), *(const v4f*)(p + 4),
               *(const v4f*)(p + 16), *(const v4f*)(p + 20));
}

__device__ __forceinline__ v16bf load_afrag_bf16(const bf16_t* __restrict__ base,
                                                 long ld, int row0, int k0) {
  int lane = threadIdx.x & 31;
  int r = lane & 15, sel = lane >> 4;
  const bf16_t* p = base + (long)(row0 + r) * ld + k0 + sel * 8;
  v8bf lo = *(const v8bf*)(p);       // K +0..7
  v8bf hi = *(const v8bf*)(p + 16);  // K +16..23
  v16bf f;
#pragma unroll
  for (int i = 0; i < 8; ++i) { f[i] = lo[i]; f[8 + i] = hi[i]; }
  return f;
}

// A-fragment from an fp32 LDS chunk buf[MROWS][KCH+4] (pad -> conflict-free)
__device__ __forceinline__ v16bf afrag_lds_f32(const float (*buf)[KCH + 4],
                                               int row0, int kk) {
  int lane = threadIdx.x & 31;
  int r = lane & 15, sel = lane >> 4;
  const float* p = &buf[row0 + r][kk + sel * 8];
  return cvt16(*(const v4f*)(p + 0), *(const v4f*)(p + 4),
               *(const v4f*)(p + 16), *(const v4f*)(p + 20));
}

// cooperative 512-thread stage of a MROWS x KCH fp32 adj chunk into LDS
// thread t copies 8 consecutive floats (2 async b128): 512*8 = 64*64 elements
__device__ __forceinline__ void stage_adj(const float* __restrict__ adj,
                                          float (*buf)[KCH + 4], long m0, int k0) {
  int t = threadIdx.x;
  int row = t >> 3;            // 0..63
  int col = (t & 7) * 8;       // 0..56
  const float* src = adj + (m0 + row) * (long)NNODES + k0 + col;
  float* dst = &buf[row][col];
#if USE_ASYNC_LDS
  __builtin_amdgcn_global_load_async_to_lds_b128(
      (as1_v4i*)(unsigned long long)(const void*)src,
      (as3_v4i*)(unsigned)(unsigned long long)(void*)dst, 0, 0);
  __builtin_amdgcn_global_load_async_to_lds_b128(
      (as1_v4i*)(unsigned long long)(const void*)(src + 4),
      (as3_v4i*)(unsigned)(unsigned long long)(void*)(dst + 4), 0, 0);
#else
  *(v4f*)dst = *(const v4f*)src;
  *(v4f*)(dst + 4) = *(const v4f*)(src + 4);
#endif
}

// out[n*K + k] = (bf16) in[k*N + n]   (W1 -> W1T, W2 -> W2T)
__global__ void k_transpose_f32_to_bf16(const float* __restrict__ in,
                                        bf16_t* __restrict__ out, int K, int Ncols) {
  int idx = blockIdx.x * blockDim.x + threadIdx.x;
  if (idx >= K * Ncols) return;
  int n = idx / K, k = idx - n * K;
  out[idx] = bf(in[(long)k * Ncols + n]);
}

// XW1T[n][m] = (x @ W1)[m][n], bf16, 256 x 16384. One wave per 16x16 tile.
__global__ __launch_bounds__(256) void k_gemm_xw1(const float* __restrict__ x,
                                                  const bf16_t* __restrict__ W1T,
                                                  bf16_t* __restrict__ XW1T) {
  int wid  = (blockIdx.x * blockDim.x + threadIdx.x) >> 5;
  int lane = threadIdx.x & 31;
  int m0 = (wid >> 4) * 16;
  int n0 = (wid & 15) * 16;
  v8f c = {};
  for (int ks = 0; ks < INDIM / 32; ++ks) {
    int k0 = ks << 5;
    v16bf a = load_afrag_f32(x, INDIM, m0, k0);
    v16bf b = load_afrag_bf16(W1T, INDIM, n0, k0);
    c = WMMA_BF16(a, b, c);
  }
  // C layout: vgpr r, lanes<16 -> (M=r,N=lane); lanes>=16 -> (M=r+8,N=lane-16).
  // Transposed store is contiguous in m: one b128 store per lane.
  int r15 = lane & 15, sel = lane >> 4;
  v8bf pk;
#pragma unroll
  for (int r = 0; r < 8; ++r) pk[r] = bf(c[r]);
  *(v8bf*)(XW1T + (long)(n0 + r15) * NNODES + m0 + sel * 8) = pk;
}

// Fused layer 1 (512 thr = 16 waves, 64-row M-tile):
//   Z1[64x256] = adj[64xN] @ XW1  (adj via async LDS pipeline, B reused 4x)
//   relu -> LDS, then HW2[64x128] = h @ W2 straight from LDS.
__global__ __launch_bounds__(512) void k_gcn_layer1(const float* __restrict__ adj,
                                                    const bf16_t* __restrict__ XW1T,
                                                    const bf16_t* __restrict__ W2T,
                                                    bf16_t* __restrict__ HW2T) {
  __shared__ float  ldsA[2][MROWS][KCH + 4];   // 2 x 17 KB adj chunks (fp32)
  __shared__ bf16_t hT[MROWS][H1DIM + 8];      // relu(Z1) strip, padded rows
  long m0  = (long)blockIdx.x * MROWS;
  int  w    = threadIdx.x >> 5;                // 0..15 -> Z1 n-tile
  int  lane = threadIdx.x & 31;
  int  r15 = lane & 15, sel = lane >> 4;
  int  n0 = w * 16;

  v8f acc[4] = {};                             // 4 M-subtiles per wave

  stage_adj(adj, ldsA[0], m0, 0);
  const int NCH = NNODES / KCH;
  for (int cs = 0; cs < NCH; ++cs) {
    int cur = cs & 1;
    if (cs + 1 < NCH) {
      stage_adj(adj, ldsA[cur ^ 1], m0, (cs + 1) * KCH);
      WAIT_ASYNC(2);
    } else {
      WAIT_ASYNC(0);
    }
    __syncthreads();
#pragma unroll
    for (int kk = 0; kk < KCH; kk += 32) {
      v16bf b = load_afrag_bf16(XW1T, NNODES, n0, cs * KCH + kk);
#pragma unroll
      for (int mt = 0; mt < 4; ++mt) {         // B held in regs across 4 WMMAs
        v16bf a = afrag_lds_f32(ldsA[cur], mt * 16, kk);
        acc[mt] = WMMA_BF16(a, b, acc[mt]);
      }
    }
    __syncthreads();
  }

#pragma unroll
  for (int mt = 0; mt < 4; ++mt)
#pragma unroll
    for (int r = 0; r < 8; ++r)
      hT[mt * 16 + r + sel * 8][n0 + r15] = bf(fmaxf(acc[mt][r], 0.0f));
  __syncthreads();

  // stage 2: waves 0-7 -> HW2 n-tile w, M-subtiles {0,1}; waves 8-15 -> {2,3}
  int n2 = (w & 7) * 16;
  int mbase = (w >> 3) * 2;
  v8f a2[2] = {};
#pragma unroll
  for (int ks = 0; ks < H1DIM / 32; ++ks) {
    int k0 = ks << 5;
    v16bf b = load_afrag_bf16(W2T, H1DIM, n2, k0);
#pragma unroll
    for (int mt = 0; mt < 2; ++mt) {
      int mrow = (mbase + mt) * 16;
      v8bf lo = *(const v8bf*)&hT[mrow + r15][k0 + sel * 8];
      v8bf hi = *(const v8bf*)&hT[mrow + r15][k0 + sel * 8 + 16];
      v16bf a;
#pragma unroll
      for (int i = 0; i < 8; ++i) { a[i] = lo[i]; a[8 + i] = hi[i]; }
      a2[mt] = WMMA_BF16(a, b, a2[mt]);
    }
  }
#pragma unroll
  for (int mt = 0; mt < 2; ++mt) {
    v8bf pk;
#pragma unroll
    for (int r = 0; r < 8; ++r) pk[r] = bf(a2[mt][r]);
    *(v8bf*)(HW2T + (long)(n2 + r15) * NNODES + m0 + (mbase + mt) * 16 + sel * 8) = pk;
  }
}

// Layer 2 (512 thr, 64-row M-tile): Z[16384x128] f32 = adj @ HW2.
// waves 0-7 -> n-tile w, M-subtiles {0,1}; waves 8-15 -> same n, {2,3}.
__global__ __launch_bounds__(512) void k_gcn_layer2(const float* __restrict__ adj,
                                                    const bf16_t* __restrict__ HW2T,
                                                    float* __restrict__ Z) {
  __shared__ float ldsA[2][MROWS][KCH + 4];
  long m0  = (long)blockIdx.x * MROWS;
  int  w    = threadIdx.x >> 5;
  int  lane = threadIdx.x & 31;
  int  r15 = lane & 15, sel = lane >> 4;
  int  n0 = (w & 7) * 16;
  int  mbase = (w >> 3) * 2;
  v8f acc[2] = {};

  stage_adj(adj, ldsA[0], m0, 0);
  const int NCH = NNODES / KCH;
  for (int cs = 0; cs < NCH; ++cs) {
    int cur = cs & 1;
    if (cs + 1 < NCH) {
      stage_adj(adj, ldsA[cur ^ 1], m0, (cs + 1) * KCH);
      WAIT_ASYNC(2);
    } else {
      WAIT_ASYNC(0);
    }
    __syncthreads();
#pragma unroll
    for (int kk = 0; kk < KCH; kk += 32) {
      v16bf b = load_afrag_bf16(HW2T, NNODES, n0, cs * KCH + kk);
#pragma unroll
      for (int mt = 0; mt < 2; ++mt) {
        v16bf a = afrag_lds_f32(ldsA[cur], (mbase + mt) * 16, kk);
        acc[mt] = WMMA_BF16(a, b, acc[mt]);
      }
    }
    __syncthreads();
  }
#pragma unroll
  for (int mt = 0; mt < 2; ++mt)
#pragma unroll
    for (int r = 0; r < 8; ++r)
      Z[(m0 + (mbase + mt) * 16 + r + sel * 8) * H2DIM + n0 + r15] = acc[mt][r];
}

// weff[i] = sum_j Wtwo[i][j] * Wthree[j],  i in [0,256)
__global__ void k_weff(const float* __restrict__ Wtwo, const float* __restrict__ Wthree,
                       float* __restrict__ weff) {
  int i = threadIdx.x;
  float s = 0.0f;
  for (int j = 0; j < H2DIM; ++j) s += Wtwo[i * H2DIM + j] * Wthree[j];
  weff[i] = s;
}

// One wave per edge: coalesced v4f gathers of L2-resident Z rows + shuffle reduce.
__global__ __launch_bounds__(256) void k_decode(const int* __restrict__ edges, int nE,
                                                const float* __restrict__ Z,
                                                const float* __restrict__ weff,
                                                const float* __restrict__ Wthree,
                                                float* __restrict__ out) {
  int wid  = (blockIdx.x * blockDim.x + threadIdx.x) >> 5;
  int lane = threadIdx.x & 31;
  if (wid >= nE) return;
  int i = edges[2 * wid];
  int j = edges[2 * wid + 1];
  const float* zi = Z + (long)i * H2DIM;
  const float* zj = Z + (long)j * H2DIM;
  int t = lane * 4;  // 32 lanes * 4 = 128
  v4f a  = *(const v4f*)(zi + t);
  v4f b  = *(const v4f*)(zj + t);
  v4f w0 = *(const v4f*)(weff + t);
  v4f w1 = *(const v4f*)(weff + H2DIM + t);
  v4f w3 = *(const v4f*)(Wthree + H2DIM + t);
  float s = 0.0f;
#pragma unroll
  for (int q = 0; q < 4; ++q)
    s += fmaxf(a[q], 0.0f) * w0[q] + fmaxf(b[q], 0.0f) * w1[q] + a[q] * b[q] * w3[q];
#pragma unroll
  for (int off = 16; off > 0; off >>= 1) s += __shfl_down(s, off, 32);
  if (lane == 0) out[wid] = 1.0f / (1.0f + __expf(-s));
}

extern "C" void kernel_launch(void* const* d_in, const int* in_sizes, int n_in,
                              void* d_out, int out_size, void* d_ws, size_t ws_size,
                              hipStream_t stream) {
  const float* x      = (const float*)d_in[0];
  const float* adj    = (const float*)d_in[1];
  const float* W1     = (const float*)d_in[2];
  const float* W2     = (const float*)d_in[3];
  const float* Wtwo   = (const float*)d_in[4];
  const float* Wthree = (const float*)d_in[5];
  const int*   e_pos  = (const int*)d_in[6];
  const int*   e_neg  = (const int*)d_in[7];
  float* out = (float*)d_out;
  const int E = in_sizes[6] / 2;

  // workspace layout (256B-aligned offsets), ~20.3 MB total
  char* ws = (char*)d_ws;
  bf16_t* W1T  = (bf16_t*)(ws + 0);         //  256*512*2   = 256 KB
  bf16_t* W2T  = (bf16_t*)(ws + 262144);    //  128*256*2   =  64 KB
  bf16_t* XW1T = (bf16_t*)(ws + 327680);    //  256*16384*2 =   8 MB
  bf16_t* HW2T = (bf16_t*)(ws + 8716288);   //  128*16384*2 =   4 MB
  float*  Z    = (float*) (ws + 12910592);  //  16384*128*4 =   8 MB
  float*  weff = (float*) (ws + 21299200);  //  256*4

  k_transpose_f32_to_bf16<<<(INDIM * H1DIM + 255) / 256, 256, 0, stream>>>(W1, W1T, INDIM, H1DIM);
  k_transpose_f32_to_bf16<<<(H1DIM * H2DIM + 255) / 256, 256, 0, stream>>>(W2, W2T, H1DIM, H2DIM);
  k_gemm_xw1<<<(NNODES / 16) * (H1DIM / 16) / 8, 256, 0, stream>>>(x, W1T, XW1T);
  k_gcn_layer1<<<NNODES / MROWS, 512, 0, stream>>>(adj, XW1T, W2T, HW2T);
  k_gcn_layer2<<<NNODES / MROWS, 512, 0, stream>>>(adj, HW2T, Z);
  k_weff<<<1, 256, 0, stream>>>(Wtwo, Wthree, weff);
  int dec_blocks = (E + 7) / 8;  // 8 waves (edges) per 256-thread block
  k_decode<<<dec_blocks, 256, 0, stream>>>(e_pos, E, Z, weff, Wthree, out);
  k_decode<<<dec_blocks, 256, 0, stream>>>(e_neg, E, Z, weff, Wthree, out + E);
}
